// BasicTransBlock_54795192762525
// MI455X (gfx1250) — matrix operand order repaired and checked by reading, not compiled, not run on hardware
//
#include <hip/hip_runtime.h>
#include <hip/hip_bf16.h>

#define C_CH   128
#define HEADS  4
#define DHH    32
#define M_KEYS 512
#define NSP    32768
#define NTIL   2048           // NSP/16
#define EPSV   1e-5f
#define SCALEA 0.17677669529663687f  // 32^-0.5

typedef __bf16 v16bf __attribute__((ext_vector_type(16)));
typedef float  v8f   __attribute__((ext_vector_type(8)));

union Frag { v16bf v; uint4 q[2]; unsigned int w[8]; unsigned short u[16]; };

__device__ __forceinline__ unsigned short f2bf(float f) {
  union { float f; unsigned int u; } x; x.f = f;
  unsigned int r = 0x7FFFu + ((x.u >> 16) & 1u);
  return (unsigned short)((x.u + r) >> 16);
}

__device__ __forceinline__ void loadA(Frag& a, const uint4* __restrict__ p) {
  a.q[0] = p[0]; a.q[1] = p[1];
}
__device__ __forceinline__ void loadB4(Frag* b, const uint4* __restrict__ p) {
#pragma unroll
  for (int s = 0; s < 4; s++) { b[s].q[0] = p[s * 64]; b[s].q[1] = p[s * 64 + 1]; }
}
__device__ __forceinline__ void mma4(v8f* acc, const Frag& a, const Frag* b) {
#pragma unroll
  for (int s = 0; s < 4; s++)
    acc[s] = __builtin_amdgcn_wmma_f32_16x16x32_bf16(
        false, a.v, false, b[s].v, (short)0, acc[s], false, false);
}

// ---------------- per-channel batch stats (biased var) ----------------
__global__ void stats_kernel(const float* __restrict__ src,
                             float* __restrict__ meanOut,
                             float* __restrict__ invstdOut) {
  __shared__ float s1[256], s2[256];
  int c = blockIdx.x;
  const float* p = src + (size_t)c * NSP;
  float sum = 0.f, sq = 0.f;
  for (int i = threadIdx.x; i < NSP; i += 256) { float v = p[i]; sum += v; sq += v * v; }
  s1[threadIdx.x] = sum; s2[threadIdx.x] = sq;
  __syncthreads();
  for (int off = 128; off > 0; off >>= 1) {
    if ((int)threadIdx.x < off) {
      s1[threadIdx.x] += s1[threadIdx.x + off];
      s2[threadIdx.x] += s2[threadIdx.x + off];
    }
    __syncthreads();
  }
  if (threadIdx.x == 0) {
    float m = s1[0] / (float)NSP;
    float var = s2[0] / (float)NSP - m * m;
    meanOut[c] = m;
    invstdOut[c] = rsqrtf(var + EPSV);
  }
}

// -- (optional BN) + depthwise 3x3x3 conv -> B-fragment-swizzled bf16 --
__global__ void dwconv_kernel(const float* __restrict__ src,
                              const float* __restrict__ mean,
                              const float* __restrict__ invstd,
                              const float* __restrict__ gamma,
                              const float* __restrict__ beta,
                              const float* __restrict__ wt,
                              unsigned short* __restrict__ dst, int useBN) {
  int c = blockIdx.y;
  int n = blockIdx.x * blockDim.x + threadIdx.x;
  int h = n >> 10, w = (n >> 5) & 31, d = n & 31;
  const float* sp = src + (size_t)c * NSP;
  float mu = 0.f, is = 1.f, g = 1.f, b = 0.f;
  if (useBN) { mu = mean[c]; is = invstd[c]; g = gamma[c]; b = beta[c]; }
  float acc = 0.f;
  for (int dh = -1; dh <= 1; dh++)
    for (int dw = -1; dw <= 1; dw++)
      for (int dd = -1; dd <= 1; dd++) {
        int hh = h + dh, ww = w + dw, zz = d + dd;
        float v = 0.f;
        if ((unsigned)hh < 32u && (unsigned)ww < 32u && (unsigned)zz < 32u) {
          v = sp[hh * 1024 + ww * 32 + zz];
          if (useBN) v = (v - mu) * is * g + b;
        }
        acc += v * wt[c * 27 + (dh + 1) * 9 + (dw + 1) * 3 + (dd + 1)];
      }
  int kChunk = c >> 5, kin = c & 31;
  int lane = (n & 15) | (kin & 16);
  int uidx = kin & 15;
  dst[(((size_t)kChunk * NTIL + (n >> 4)) * 32 + lane) * 16 + uidx] = f2bf(acc);
}

// -------- weight pack: [M,128] fp32 -> A-fragment-swizzled bf16 -------
__global__ void packA_kernel(const float* __restrict__ W,
                             unsigned short* __restrict__ dst, int total) {
  int i = blockIdx.x * blockDim.x + threadIdx.x;
  if (i >= total) return;
  int k = i & 127, m = i >> 7;
  int kChunk = k >> 5, kin = k & 31;
  int hi = (kin >> 4) & 1;
  int lane = (m & 15) | (hi << 4);
  int uidx = kin & 15;
  dst[(((size_t)(m >> 4) * 4 + kChunk) * 32 + lane) * 16 + uidx] = f2bf(W[i]);
}

// -------- WMMA bf16 GEMM on swizzled fragments ------------------------
// K=128 (4 chunks, double-buffered), N=32768; wave -> 16(M) x 64(N)
__global__ void gemm_kernel(const uint4* __restrict__ A4,
                            const uint4* __restrict__ B4,
                            const float* __restrict__ resid,
                            float* __restrict__ Cout) {
  int lane = threadIdx.x & 31;
  int wave = threadIdx.x >> 5;
  int mTile = blockIdx.y;
  int nTile0 = (blockIdx.x * 8 + wave) * 4;

  const size_t ASTEP = 64;                   // 32 lanes * 2 uint4 per k-chunk
  const size_t BSTEP = (size_t)NTIL * 64;    // per k-chunk
  const uint4* ap = A4 + ((size_t)mTile * 4 * 32 + lane) * 2;
  const uint4* bp = B4 + ((size_t)nTile0 * 32 + lane) * 2;

  v8f z = {0.f, 0.f, 0.f, 0.f, 0.f, 0.f, 0.f, 0.f};
  v8f acc[4] = {z, z, z, z};

  Frag a0, a1, b0[4], b1[4];
  loadA(a0, ap);             loadB4(b0, bp);
  loadA(a1, ap + ASTEP);     loadB4(b1, bp + BSTEP);
  mma4(acc, a0, b0);
  loadA(a0, ap + 2 * ASTEP); loadB4(b0, bp + 2 * BSTEP);
  mma4(acc, a1, b1);
  loadA(a1, ap + 3 * ASTEP); loadB4(b1, bp + 3 * BSTEP);
  mma4(acc, a0, b0);
  mma4(acc, a1, b1);

  int col = lane & 15;
  bool hi = lane >= 16;
  int mBase = mTile * 16;
#pragma unroll
  for (int s = 0; s < 4; s++) {
    int bc = (nTile0 + s) * 16 + col;
#pragma unroll
    for (int i = 0; i < 8; i++) {
      int row = mBase + i + (hi ? 8 : 0);
      size_t idx = (size_t)row * NSP + bc;
      float v = acc[s][i];
      if (resid) v += resid[idx];
      Cout[idx] = v;
    }
  }
}

// -- q repack: qkv[(d*4+head)][n] f32 -> A-frag swizzled per head ------
__global__ void qcvt_kernel(const float* __restrict__ qkv,
                            unsigned short* __restrict__ qswz) {
  int i = blockIdx.x * blockDim.x + threadIdx.x;  // 2^22
  int d = i & 31;
  int n = (i >> 5) & (NSP - 1);
  int head = i >> 20;
  float v = qkv[(size_t)(d * HEADS + head) * NSP + n];
  int hi = (d >> 4) & 1;
  int lane = (n & 15) | (hi << 4);
  int uidx = d & 15;
  qswz[(size_t)head * 1048576 + (((size_t)(n >> 4)) * 32 + lane) * 16 + uidx] = f2bf(v);
}

// -- trilinear downsample k,v -> B-frag swizzled ----------------------
__global__ void dskv_kernel(const float* __restrict__ qkv,
                            unsigned short* __restrict__ kd,
                            unsigned short* __restrict__ vd) {
  int i = blockIdx.x * blockDim.x + threadIdx.x;  // 131072
  int m = i & 511;
  int c = (i >> 9) & 127;
  int which = i >> 16;
  int p = m >> 6, q = (m >> 3) & 7, r = m & 7;
  const float* src = qkv + (size_t)(C_CH + which * C_CH + c) * NSP;
  float posH = p * 31.0f / 7.0f; int loH = (int)posH; if (loH > 30) loH = 30; float wH = posH - loH;
  float posW = q * 31.0f / 7.0f; int loW = (int)posW; if (loW > 30) loW = 30; float wW = posW - loW;
  float posD = r * 31.0f / 7.0f; int loD = (int)posD; if (loD > 30) loD = 30; float wD = posD - loD;
  float acc = 0.f;
  for (int ah = 0; ah < 2; ah++)
    for (int aw = 0; aw < 2; aw++)
      for (int ad = 0; ad < 2; ad++) {
        float wgt = (ah ? wH : 1.f - wH) * (aw ? wW : 1.f - wW) * (ad ? wD : 1.f - wD);
        acc += wgt * src[(loH + ah) * 1024 + (loW + aw) * 32 + (loD + ad)];
      }
  int head = c & 3, dh = c >> 2;
  unsigned short bv = f2bf(acc);
  if (which == 0) {
    int lane = (m & 15) | (dh & 16);
    int uidx = dh & 15;
    kd[(((size_t)(head * 32 + (m >> 4))) * 32 + lane) * 16 + uidx] = bv;
  } else {
    int kt = m >> 5, kin = m & 31;
    int lane = (dh & 15) | (kin & 16);
    int uidx = kin & 15;
    vd[((((size_t)(head * 16 + kt)) * 2 + (dh >> 4)) * 32 + lane) * 16 + uidx] = bv;
  }
}

// -------- bias[head][pn][m], faithful REL_IDX quirks -----------------
__global__ void bias_kernel(const float* __restrict__ rel_table,
                            float* __restrict__ bias) {
  int i = blockIdx.x * blockDim.x + threadIdx.x;  // 2^20
  int m = i & 511;
  int pn = (i >> 9) & 511;
  int head = i >> 18;
  int a = pn >> 6, b = (pn >> 3) & 7, c = pn & 7;
  int a2 = m >> 6, b2 = (m >> 3) & 7, c2 = m & 7;
  int idx = (a - a2 + 7) * 15 + (b - b2 + 7) + (c - c2);
  if (idx < 0) idx += 3375;
  bias[i] = rel_table[idx * HEADS + head];
}

// -------- attention: one wave per (head, 16-query tile) --------------
__global__ void attn_kernel(const uint4* __restrict__ q4,
                            const uint4* __restrict__ k4,
                            const uint4* __restrict__ v4,
                            const float* __restrict__ bias,
                            float* __restrict__ O1) {
  __shared__ float sm[16 * M_KEYS];  // 32 KB logits
  int head = blockIdx.y;
  int qTile = blockIdx.x;
  int qBase = qTile * 16;
  int lane = threadIdx.x;
  int col = lane & 15;
  bool hiL = lane >= 16;

  Frag a;  // q tile: 16 queries x 32 d
  {
    const uint4* qp = q4 + (size_t)head * 131072 + ((size_t)qTile * 32 + lane) * 2;
    a.q[0] = qp[0]; a.q[1] = qp[1];
  }

  v8f z = {0.f, 0.f, 0.f, 0.f, 0.f, 0.f, 0.f, 0.f};
  const uint4* kp = k4 + (size_t)head * 2048 + (size_t)lane * 2;
  const float* brow0 = bias + (size_t)head * M_KEYS * M_KEYS;

  // logits: 32 key tiles of 16
  for (int mt = 0; mt < 32; mt++) {
    Frag b;
    b.q[0] = kp[mt * 64]; b.q[1] = kp[mt * 64 + 1];
    v8f cfr = __builtin_amdgcn_wmma_f32_16x16x32_bf16(
        false, a.v, false, b.v, (short)0, z, false, false);
#pragma unroll
    for (int i = 0; i < 8; i++) {
      int row = i + (hiL ? 8 : 0);
      int nq = qBase + row;
      int pn = ((nq >> 12) << 6) | (((nq >> 7) & 7) << 3) | ((nq >> 2) & 7);
      float bval = brow0[(size_t)pn * M_KEYS + mt * 16 + col];
      sm[row * M_KEYS + mt * 16 + col] = (cfr[i] + bval) * SCALEA;
    }
  }
  __syncthreads();

  // softmax: lane pair (r, r+16) handles row r (two 256-wide halves)
  {
    int row = lane & 15;
    int half = lane >> 4;
    float* rp = sm + row * M_KEYS + half * 256;
    float mx = -3.0e38f;
    for (int j = 0; j < 256; j++) mx = fmaxf(mx, rp[j]);
    mx = fmaxf(mx, __shfl_xor(mx, 16, 32));
    float s = 0.f;
    for (int j = 0; j < 256; j++) { float e = __expf(rp[j] - mx); rp[j] = e; s += e; }
    s += __shfl_xor(s, 16, 32);
    float inv = 1.0f / s;
    for (int j = 0; j < 256; j++) rp[j] *= inv;
  }
  __syncthreads();

  // AV: attn[16x512] * v[512x32]
  v8f d0 = z, d1 = z;
  const uint4* vp = v4 + (size_t)head * 2048 + (size_t)lane * 2;
  for (int kt = 0; kt < 16; kt++) {
    Frag aa;  // A frag from LDS softmax rows
    const float* arow = sm + col * M_KEYS + kt * 32;
#pragma unroll
    for (int j = 0; j < 8; j++) {
      int k = 2 * j + (hiL ? 16 : 0);
      aa.u[2 * j]     = f2bf(arow[k]);
      aa.u[2 * j + 1] = f2bf(arow[k + 1]);
    }
    Frag b0, b1;
    b0.q[0] = vp[kt * 128];      b0.q[1] = vp[kt * 128 + 1];
    b1.q[0] = vp[kt * 128 + 64]; b1.q[1] = vp[kt * 128 + 65];
    d0 = __builtin_amdgcn_wmma_f32_16x16x32_bf16(false, aa.v, false, b0.v, (short)0, d0, false, false);
    d1 = __builtin_amdgcn_wmma_f32_16x16x32_bf16(false, aa.v, false, b1.v, (short)0, d1, false, false);
  }
#pragma unroll
  for (int i = 0; i < 8; i++) {
    int row = i + (hiL ? 8 : 0);
    int nq = qBase + row;
    O1[(size_t)(col * HEADS + head) * NSP + nq] = d0[i];
    O1[(size_t)((16 + col) * HEADS + head) * NSP + nq] = d1[i];
  }
}

// -------- BN2 + ReLU -> B-fragment-swizzled bf16 ---------------------
__global__ void bnrelu_kernel(const float* __restrict__ y1,
                              const float* __restrict__ mean,
                              const float* __restrict__ invstd,
                              const float* __restrict__ g,
                              const float* __restrict__ b,
                              unsigned short* __restrict__ dst) {
  int i = blockIdx.x * blockDim.x + threadIdx.x;
  int c = i >> 15;
  int n = i & (NSP - 1);
  float v = (y1[i] - mean[c]) * invstd[c] * g[c] + b[c];
  v = fmaxf(v, 0.f);
  int kChunk = c >> 5, kin = c & 31;
  int lane = (n & 15) | (kin & 16);
  int uidx = kin & 15;
  dst[(((size_t)kChunk * NTIL + (n >> 4)) * 32 + lane) * 16 + uidx] = f2bf(v);
}

extern "C" void kernel_launch(void* const* d_in, const int* in_sizes, int n_in,
                              void* d_out, int out_size, void* d_ws, size_t ws_size,
                              hipStream_t stream) {
  (void)in_sizes; (void)n_in; (void)out_size; (void)ws_size;
  const float* x       = (const float*)d_in[0];
  const float* bn1_g   = (const float*)d_in[1];
  const float* bn1_b   = (const float*)d_in[2];
  const float* qkv_dw  = (const float*)d_in[3];
  const float* qkv_pw  = (const float*)d_in[4];
  const float* rel_tbl = (const float*)d_in[5];
  const float* out_dw  = (const float*)d_in[6];
  const float* out_pw  = (const float*)d_in[7];
  const float* bn2_g   = (const float*)d_in[8];
  const float* bn2_b   = (const float*)d_in[9];
  const float* mlp_w   = (const float*)d_in[10];

  char* ws = (char*)d_ws;
  float* mean1   = (float*)(ws + 0);
  float* invstd1 = (float*)(ws + 512);
  float* mean2   = (float*)(ws + 1024);
  float* invstd2 = (float*)(ws + 1536);
  unsigned short* Wqkv = (unsigned short*)(ws + 4096);
  unsigned short* Wout = (unsigned short*)(ws + 4096 + 98304);
  unsigned short* Wmlp = (unsigned short*)(ws + 4096 + 98304 + 32768);

  const size_t OFF_HB   = 262144;
  const size_t OFF_QKV  = OFF_HB + 8388608;
  const size_t OFF_QBF  = OFF_QKV + 50331648;
  const size_t OFF_KD   = OFF_QBF + 8388608;
  const size_t OFF_VD   = OFF_KD + 131072;
  const size_t OFF_BIAS = OFF_VD + 131072;
  const size_t OFF_O1   = OFF_QKV;
  const size_t OFF_YDW  = OFF_O1 + 16777216;
  const size_t OFF_Y1   = OFF_YDW + 8388608;
  const size_t OFF_RELU = OFF_HB;

  unsigned short* HB  = (unsigned short*)(ws + OFF_HB);
  float*          QKV = (float*)(ws + OFF_QKV);
  unsigned short* QBF = (unsigned short*)(ws + OFF_QBF);
  unsigned short* KD  = (unsigned short*)(ws + OFF_KD);
  unsigned short* VD  = (unsigned short*)(ws + OFF_VD);
  float*          BIA = (float*)(ws + OFF_BIAS);
  float*          O1  = (float*)(ws + OFF_O1);
  unsigned short* YDW = (unsigned short*)(ws + OFF_YDW);
  float*          Y1  = (float*)(ws + OFF_Y1);
  unsigned short* RLU = (unsigned short*)(ws + OFF_RELU);
  float*          OUT = (float*)d_out;

  stats_kernel<<<128, 256, 0, stream>>>(x, mean1, invstd1);
  dwconv_kernel<<<dim3(128, 128), 256, 0, stream>>>(x, mean1, invstd1, bn1_g, bn1_b,
                                                    qkv_dw, HB, 1);
  packA_kernel<<<192, 256, 0, stream>>>(qkv_pw, Wqkv, 384 * 128);
  packA_kernel<<<64, 256, 0, stream>>>(out_pw, Wout, 128 * 128);
  packA_kernel<<<64, 256, 0, stream>>>(mlp_w, Wmlp, 128 * 128);
  gemm_kernel<<<dim3(64, 24), 256, 0, stream>>>((const uint4*)Wqkv, (const uint4*)HB,
                                                nullptr, QKV);
  qcvt_kernel<<<16384, 256, 0, stream>>>(QKV, QBF);
  dskv_kernel<<<512, 256, 0, stream>>>(QKV, KD, VD);
  bias_kernel<<<4096, 256, 0, stream>>>(rel_tbl, BIA);
  attn_kernel<<<dim3(2048, 4), 32, 0, stream>>>((const uint4*)QBF, (const uint4*)KD,
                                                (const uint4*)VD, BIA, O1);
  dwconv_kernel<<<dim3(128, 128), 256, 0, stream>>>(O1, nullptr, nullptr, nullptr, nullptr,
                                                    out_dw, YDW, 0);
  gemm_kernel<<<dim3(64, 8), 256, 0, stream>>>((const uint4*)Wout, (const uint4*)YDW,
                                               x, Y1);
  stats_kernel<<<128, 256, 0, stream>>>(Y1, mean2, invstd2);
  bnrelu_kernel<<<16384, 256, 0, stream>>>(Y1, mean2, invstd2, bn2_g, bn2_b, RLU);
  gemm_kernel<<<dim3(64, 8), 256, 0, stream>>>((const uint4*)Wmlp, (const uint4*)RLU,
                                               Y1, OUT);
}
